// DeformConv_50637664419876
// MI455X (gfx1250) — matrix operand order repaired and checked by reading, not compiled, not run on hardware
//
#include <hip/hip_runtime.h>
#include <hip/hip_bf16.h>

typedef __attribute__((ext_vector_type(16))) __bf16 v16bf;
typedef __attribute__((ext_vector_type(8)))  float  v8f;

namespace {
constexpr int kC = 256, kHW = 64 * 64;
constexpr int kKdim = 9 * kC; // 2304
}

// ---------------------------------------------------------------------------
// Repack weight (O,C,3,3) fp32 -> bf16 B^T laid out as Bw[o][kk*256 + c]
// so that the GEMM reduction order is (kk, c) with c contiguous.
// ---------------------------------------------------------------------------
__global__ void repack_weight_bf16(const float* __restrict__ w,
                                   __bf16* __restrict__ bw) {
  int idx = blockIdx.x * blockDim.x + threadIdx.x; // 0 .. 256*2304-1
  int o  = idx / kKdim;
  int r  = idx - o * kKdim;
  int kk = r >> 8;
  int c  = r & 255;
  bw[idx] = (__bf16)w[o * kKdim + c * 9 + kk];
}

// ---------------------------------------------------------------------------
// Fused deformable-im2col + bf16 WMMA GEMM.
// Grid: 256 blocks (one per (b,h) output row of 64 pixels), 256 threads.
// Each block computes out[b, 0:256, h, 0:64].
// ---------------------------------------------------------------------------
__launch_bounds__(256)
__global__ void deform_conv_wmma(const float* __restrict__ x,
                                 const float* __restrict__ offs,
                                 const float* __restrict__ msk,
                                 const __bf16* __restrict__ bw,
                                 float* __restrict__ out) {
  __shared__ __align__(16) __bf16 s_a[64 * 32];   // A tile: 64 pixels x 32 k
  __shared__ __align__(16) __bf16 s_b[256 * 32];  // B tile: 256 outs x 32 k
  __shared__ int   s_off[64][4];
  __shared__ float s_wgt[64][4];

  const int tid  = threadIdx.x;
  const int blk  = blockIdx.x;      // 0..255
  const int b    = blk >> 6;        // batch
  const int h    = blk & 63;        // output row

  const int lane   = tid & 31;
  const int wv     = tid >> 5;      // 0..7
  const int m_sub  = wv >> 1;       // 0..3 : which 16 pixels
  const int n_hlf  = wv & 1;        // 0..1 : which 128 output channels
  const int nrow   = lane & 15;
  const int hlf    = lane >> 4;

  const float* xb = x + (size_t)b * kC * kHW;

  v8f acc[8];
  const v8f vzero = {0.f, 0.f, 0.f, 0.f, 0.f, 0.f, 0.f, 0.f};
#pragma unroll
  for (int j = 0; j < 8; ++j) acc[j] = vzero;

  for (int kk = 0; kk < 9; ++kk) {
    // ---- per-kk bilinear metadata for the 64 pixels of this row ----
    if (tid < 64) {
      const int wpx = tid;
      const int ki = kk / 3, kj = kk % 3;
      const float dy = offs[(((b * 18) + kk * 2 + 0) * 64 + h) * 64 + wpx];
      const float dx = offs[(((b * 18) + kk * 2 + 1) * 64 + h) * 64 + wpx];
      const float mm = msk[(((b * 9) + kk) * 64 + h) * 64 + wpx];
      const float py = (float)(h - 1 + ki) + dy;
      const float px = (float)(wpx - 1 + kj) + dx;
      const float fy = floorf(py), fx = floorf(px);
      const int y0 = (int)fy, x0 = (int)fx;
      const float wy = py - fy, wx = px - fx;
      const int y0c = min(max(y0, 0), 63),     y1c = min(max(y0 + 1, 0), 63);
      const int x0c = min(max(x0, 0), 63),     x1c = min(max(x0 + 1, 0), 63);
      const float vy0 = (y0 >= 0 && y0 < 64) ? 1.f : 0.f;
      const float vy1 = (y0 + 1 >= 0 && y0 + 1 < 64) ? 1.f : 0.f;
      const float vx0 = (x0 >= 0 && x0 < 64) ? 1.f : 0.f;
      const float vx1 = (x0 + 1 >= 0 && x0 + 1 < 64) ? 1.f : 0.f;
      s_off[wpx][0] = y0c * 64 + x0c;
      s_off[wpx][1] = y0c * 64 + x1c;
      s_off[wpx][2] = y1c * 64 + x0c;
      s_off[wpx][3] = y1c * 64 + x1c;
      s_wgt[wpx][0] = (1.f - wy) * (1.f - wx) * vy0 * vx0 * mm;
      s_wgt[wpx][1] = (1.f - wy) * wx         * vy0 * vx1 * mm;
      s_wgt[wpx][2] = wy         * (1.f - wx) * vy1 * vx0 * mm;
      s_wgt[wpx][3] = wy         * wx         * vy1 * vx1 * mm;
    }
    __syncthreads();

    for (int cb = 0; cb < 8; ++cb) {
      const int kbase = kk * 256 + cb * 32; // K offset inside Bw rows

      // ---- build A tile: 64 pixels x 32 channels (bilinear + mask) ----
      {
        const int p = tid & 63;
        const int g = tid >> 6; // 0..3
        const int o0 = s_off[p][0], o1 = s_off[p][1];
        const int o2 = s_off[p][2], o3 = s_off[p][3];
        const float w0 = s_wgt[p][0], w1 = s_wgt[p][1];
        const float w2 = s_wgt[p][2], w3 = s_wgt[p][3];
#pragma unroll
        for (int j = 0; j < 8; ++j) {
          const int cc = g + j * 4;
          const float* xc = xb + (size_t)(cb * 32 + cc) * kHW;
          const float v = w0 * xc[o0] + w1 * xc[o1] + w2 * xc[o2] + w3 * xc[o3];
          s_a[p * 32 + cc] = (__bf16)v;
        }
      }
      // ---- stage B tile: 256 rows x 32 k, 64B contiguous per row ----
      {
        const __bf16* src = bw + (size_t)tid * kKdim + kbase;
        float4* dst = (float4*)&s_b[tid * 32];
        const float4* s4 = (const float4*)src;
        dst[0] = s4[0]; dst[1] = s4[1]; dst[2] = s4[2]; dst[3] = s4[3];
      }
      __syncthreads();

      // ---- A fragment (ISA 16-bit A layout: two b128 LDS reads) ----
      union { v16bf v; float4 q[2]; } afrag;
      const int arow = m_sub * 16 + nrow;
      afrag.q[0] = *(const float4*)(&s_a[arow * 32 + hlf * 8]);
      afrag.q[1] = *(const float4*)(&s_a[arow * 32 + 16 + hlf * 8]);

#pragma unroll
      for (int j = 0; j < 8; ++j) {
        const int orow = n_hlf * 128 + j * 16 + nrow;
        union { v16bf v; float4 q[2]; } bfrag;
        const __bf16* bp = &s_b[orow * 32 + hlf * 8];
        bfrag.q[0] = *(const float4*)(bp);
        bfrag.q[1] = *(const float4*)(bp + 16);
        acc[j] = __builtin_amdgcn_wmma_f32_16x16x32_bf16(
            false, afrag.v, false, bfrag.v, (short)0, acc[j], false, false);
      }
      __syncthreads();
    }
  }

  // ---- epilogue: C layout = (VGPR i -> row i + 8*hlf, lane%16 -> col) ----
#pragma unroll
  for (int j = 0; j < 8; ++j) {
    const int n = n_hlf * 128 + j * 16 + nrow;
    float* op = out + ((size_t)(b * 256 + n) * kHW) + h * 64;
#pragma unroll
    for (int i = 0; i < 8; ++i) {
      const int p = m_sub * 16 + hlf * 8 + i; // pixel index == w coordinate
      op[p] = acc[j][i];
    }
  }
}

extern "C" void kernel_launch(void* const* d_in, const int* in_sizes, int n_in,
                              void* d_out, int out_size, void* d_ws, size_t ws_size,
                              hipStream_t stream) {
  const float* x    = (const float*)d_in[0];
  const float* offs = (const float*)d_in[1];
  const float* msk  = (const float*)d_in[2];
  const float* wgt  = (const float*)d_in[3];
  float* out        = (float*)d_out;
  __bf16* bw        = (__bf16*)d_ws; // 256*2304*2 = 1.18 MB

  // weight -> bf16, (kk,c) K-order
  repack_weight_bf16<<<(256 * kKdim) / 256, 256, 0, stream>>>(wgt, bw);
  // fused deformable im2col + WMMA GEMM
  deform_conv_wmma<<<256, 256, 0, stream>>>(x, offs, msk, bw, out);
}
